// MultiHeadLatentAttention_25099788878444
// MI455X (gfx1250) — compile-verified, matrix operand order checked by reading, not run on hardware
//
#include <hip/hip_runtime.h>
#include <hip/hip_bf16.h>

#define DIMSZ 2048
#define NHEAD 16
#define HD    128
#define LAT   512
#define BB    2
#define SS    2048
#define ROWS  (BB*SS)   // 4096

typedef __attribute__((ext_vector_type(16))) __bf16 v16bf;
typedef __attribute__((ext_vector_type(8)))  float  v8f;

union Frag { v16bf v; uint4 u[2]; };

__device__ __forceinline__ v8f v8f_zero() {
    v8f z = {0.f,0.f,0.f,0.f,0.f,0.f,0.f,0.f};
    return z;
}

__device__ __forceinline__ v8f wmma_bf16(const Frag& a, const Frag& b, v8f c) {
    return __builtin_amdgcn_wmma_f32_16x16x32_bf16(false, a.v, false, b.v,
                                                   (short)0, c, false, false);
}

// generic pointer to a __shared__ object is aperture-form: low 32 bits = LDS offset
__device__ __forceinline__ unsigned lds_off(const void* p) {
    return (unsigned)(unsigned long long)p;
}

// async copy of 32 contiguous bytes (2 x b128), global -> LDS, per lane
__device__ __forceinline__ void async_cp32(unsigned ldsaddr, const void* gaddr) {
    unsigned long long ga = (unsigned long long)gaddr;
    asm volatile(
        "global_load_async_to_lds_b128 %0, %1, off\n\t"
        "global_load_async_to_lds_b128 %0, %1, off offset:16"
        :: "v"(ldsaddr), "v"(ga) : "memory");
}

// ---------------------------------------------------------------------------
// elementwise converts
// ---------------------------------------------------------------------------
__global__ void f32_to_bf16_kernel(const float* __restrict__ in,
                                   __bf16* __restrict__ out, int n) {
    int i = blockIdx.x * blockDim.x + threadIdx.x;
    if (i < n) out[i] = (__bf16)in[i];
}

// Wt[n*K + k] = bf16(W[k*N + n])
__global__ void transpose_to_bf16_kernel(const float* __restrict__ W,
                                         __bf16* __restrict__ Wt, int K, int N) {
    int i = blockIdx.x * blockDim.x + threadIdx.x;
    if (i < K * N) {
        int n = i / K, k = i - n * K;
        Wt[(size_t)n * K + k] = (__bf16)W[(size_t)k * N + n];
    }
}

// ---------------------------------------------------------------------------
// GEMM: C[M,N] = A[M,K](bf16,row) @ Wt[N,K](bf16)^T + bias(f32)
// Triple-buffered async LDS pipeline (GLOBAL_LOAD_ASYNC_TO_LDS_B128 /
// ASYNCcnt), one barrier per k-step, two strips in flight, buffer phase
// fully constant-folded (requires nk = K/32 with nk % 3 == 1; K in {512,2048}).
// mode 0: bf16 [M,N]    mode 1: bf16 qk-cache [bh][s][256] (features 0..127)
// mode 2: bf16 vT [bh][128][S]   mode 3: f32 [M,N]
// ---------------------------------------------------------------------------
__global__ __launch_bounds__(256)
void gemm_bf16_kernel(const __bf16* __restrict__ A, const __bf16* __restrict__ Wt,
                      const float* __restrict__ bias, void* __restrict__ out,
                      int M, int N, int K, int mode) {
    __shared__ __bf16 As[3][128][40];
    __shared__ __bf16 Bs[3][128][40];
    const int tid = threadIdx.x;
    const int wave = tid >> 5, lane = tid & 31;
    const int lanelo = lane & 15;
    const int half8 = (lane >> 4) * 8;
    const int wm = wave & 3, wn = wave >> 2;   // 4 x 2 wave grid
    const int bm = blockIdx.y * 128, bn = blockIdx.x * 128;

    v8f acc[2][4];
#pragma unroll
    for (int i = 0; i < 2; i++)
#pragma unroll
        for (int j = 0; j < 4; j++) acc[i][j] = v8f_zero();

    const int lrow = tid >> 1;        // 0..127
    const int lkp  = (tid & 1) * 16;  // 0 / 16

    const __bf16* gA = A  + (size_t)(bm + lrow) * K + lkp;
    const __bf16* gB = Wt + (size_t)(bn + lrow) * K + lkp;
    const unsigned ldsA[3] = { lds_off(&As[0][lrow][lkp]), lds_off(&As[1][lrow][lkp]),
                               lds_off(&As[2][lrow][lkp]) };
    const unsigned ldsB[3] = { lds_off(&Bs[0][lrow][lkp]), lds_off(&Bs[1][lrow][lkp]),
                               lds_off(&Bs[2][lrow][lkp]) };

    auto compute = [&](int p) {
        const int kb = half8;   // 0 or 8
        Frag af[2], bf[4];
#pragma unroll
        for (int i = 0; i < 2; i++) {
            int r = wm * 32 + i * 16 + lanelo;
            af[i].u[0] = *(const uint4*)&As[p][r][kb];
            af[i].u[1] = *(const uint4*)&As[p][r][kb + 16];
        }
#pragma unroll
        for (int j = 0; j < 4; j++) {
            int c = wn * 64 + j * 16 + lanelo;
            bf[j].u[0] = *(const uint4*)&Bs[p][c][kb];
            bf[j].u[1] = *(const uint4*)&Bs[p][c][kb + 16];
        }
#pragma unroll
        for (int i = 0; i < 2; i++)
#pragma unroll
            for (int j = 0; j < 4; j++)
                acc[i][j] = wmma_bf16(af[i], bf[j], acc[i][j]);
    };

    const __bf16* pA = gA + 64;
    const __bf16* pB = gB + 64;
    auto steady = [&](int p, int pn) {   // p, pn always literal constants
        asm volatile("s_wait_asynccnt 0x4" ::: "memory");
        __syncthreads();                 // strip p visible to all waves
        async_cp32(ldsA[pn], pA);        // pn last read in previous step: safe after barrier
        async_cp32(ldsB[pn], pB);
        pA += 32; pB += 32;
        compute(p);
    };

    const int nk = K >> 5;   // 16 or 64; nk % 3 == 1
    // prologue: two strips in flight
    async_cp32(ldsA[0], gA);
    async_cp32(ldsB[0], gB);
    async_cp32(ldsA[1], gA + 32);
    async_cp32(ldsB[1], gB + 32);

    // steady state: (nk-4)/3 full rotations + 2 steps; phases are literals
    for (int t = 0; t < (nk - 4) / 3; t++) {
        steady(0, 2);
        steady(1, 0);
        steady(2, 1);
    }
    steady(0, 2);
    steady(1, 0);
    // tails: phases 2 then 0
    asm volatile("s_wait_asynccnt 0x4" ::: "memory");
    __syncthreads();
    compute(2);
    asm volatile("s_wait_asynccnt 0x0" ::: "memory");
    __syncthreads();
    compute(0);

    // epilogue: lane holds col n = lanelo, rows r + half8 in acc VGPR r
#pragma unroll
    for (int i = 0; i < 2; i++) {
        int grow0 = bm + wm * 32 + i * 16 + half8;
#pragma unroll
        for (int j = 0; j < 4; j++) {
            int gcol = bn + wn * 64 + j * 16 + lanelo;
            float bv = bias[gcol];
#pragma unroll
            for (int r = 0; r < 8; r++) {
                float v = acc[i][j][r] + bv;
                int grow = grow0 + r;
                if (mode == 0) {
                    ((__bf16*)out)[(size_t)grow * N + gcol] = (__bf16)v;
                } else if (mode == 3) {
                    ((float*)out)[(size_t)grow * N + gcol] = v;
                } else {
                    int b = grow >> 11, s = grow & (SS - 1);
                    int h = gcol >> 7, f = gcol & (HD - 1);
                    int bh = b * NHEAD + h;
                    if (mode == 1)
                        ((__bf16*)out)[((size_t)bh * SS + s) * 256 + f] = (__bf16)v;
                    else
                        ((__bf16*)out)[((size_t)bh * HD + f) * SS + s] = (__bf16)v;
                }
            }
        }
    }
}

// ---------------------------------------------------------------------------
// RoPE elementwise: cache[bh][s][128+d] = lin[b*S+s][d] * pe(s,d), all heads
// ---------------------------------------------------------------------------
__global__ void rope_bcast_kernel(const __bf16* __restrict__ lin,
                                  __bf16* __restrict__ cache) {
    int idx = blockIdx.x * blockDim.x + threadIdx.x;
    if (idx >= ROWS * HD) return;
    int m = idx >> 7, d = idx & 127;
    int b = m >> 11, s = m & (SS - 1);
    int i2 = d & ~1;
    float freq = __expf((float)i2 * (-0.07195578415606394f)); // -ln(10000)/128
    float ang = (float)s * freq;
    float pe = (d & 1) ? __cosf(ang) : __sinf(ang);
    float v = (float)lin[idx] * pe;
#pragma unroll
    for (int h = 0; h < NHEAD; h++) {
        int bh = b * NHEAD + h;
        cache[((size_t)bh * SS + s) * 256 + 128 + d] = (__bf16)v;
    }
}

// ---------------------------------------------------------------------------
// Flash attention: Q,K caches [bh][S][256] bf16, Vt [bh][128][S] bf16
// ctx bf16 [B*S][DIM]. One wave = 16 query rows; block = 8 waves = 128 rows.
// ---------------------------------------------------------------------------
__global__ __launch_bounds__(256)
void mla_attn_kernel(const __bf16* __restrict__ Q, const __bf16* __restrict__ Kc,
                     const __bf16* __restrict__ Vt, __bf16* __restrict__ ctx) {
    __shared__ __bf16 Pl[8][16][40];
    const int wave = threadIdx.x >> 5, lane = threadIdx.x & 31;
    const int lanelo = lane & 15;
    const int half8 = (lane >> 4) * 8;
    const int kb = half8;                       // K chunk base (0/8)
    const int qblk = blockIdx.x, h = blockIdx.y, b = blockIdx.z;
    const int bh = b * NHEAD + h;
    const int q0 = qblk * 128 + wave * 16;

    const __bf16* Qb = Q  + (size_t)bh * SS * 256;
    const __bf16* Kb = Kc + (size_t)bh * SS * 256;
    const __bf16* Vb = Vt + (size_t)bh * HD * SS;

    Frag qf[8];
#pragma unroll
    for (int ks = 0; ks < 8; ks++) {
        const __bf16* p = Qb + (size_t)(q0 + lanelo) * 256 + ks * 32 + kb;
        qf[ks].u[0] = *(const uint4*)p;
        qf[ks].u[1] = *(const uint4*)(p + 16);
    }

    v8f o[8];
#pragma unroll
    for (int j = 0; j < 8; j++) o[j] = v8f_zero();
    float mi[8], li[8];
#pragma unroll
    for (int r = 0; r < 8; r++) { mi[r] = -1e30f; li[r] = 0.0f; }
    const float scale = 0.08838834764831845f;   // 1/sqrt(128)

    for (int kt = 0; kt < SS / 32; kt++) {
        const int key0 = kt * 32;
        v8f s0 = v8f_zero(), s1 = v8f_zero();
#pragma unroll
        for (int ks = 0; ks < 8; ks++) {
            Frag k0, k1;
            const __bf16* p0 = Kb + (size_t)(key0 + lanelo) * 256 + ks * 32 + kb;
            const __bf16* p1 = Kb + (size_t)(key0 + 16 + lanelo) * 256 + ks * 32 + kb;
            k0.u[0] = *(const uint4*)p0; k0.u[1] = *(const uint4*)(p0 + 16);
            k1.u[0] = *(const uint4*)p1; k1.u[1] = *(const uint4*)(p1 + 16);
            s0 = wmma_bf16(qf[ks], k0, s0);
            s1 = wmma_bf16(qf[ks], k1, s1);
        }

        // online softmax (per-row stats via half-group shuffle reduce)
        float p0v[8], p1v[8], rm[8];
#pragma unroll
        for (int r = 0; r < 8; r++) {
            float a = s0[r] * scale, c = s1[r] * scale;
            s0[r] = a; s1[r] = c;
            float mx = fmaxf(a, c);
            mx = fmaxf(mx, __shfl_xor(mx, 1, 32));
            mx = fmaxf(mx, __shfl_xor(mx, 2, 32));
            mx = fmaxf(mx, __shfl_xor(mx, 4, 32));
            mx = fmaxf(mx, __shfl_xor(mx, 8, 32));
            rm[r] = mx;
        }
#pragma unroll
        for (int r = 0; r < 8; r++) {
            float mnew  = fmaxf(mi[r], rm[r]);
            float alpha = __expf(mi[r] - mnew);
            float e0 = __expf(s0[r] - mnew);
            float e1 = __expf(s1[r] - mnew);
            float sum = e0 + e1;
            sum += __shfl_xor(sum, 1, 32);
            sum += __shfl_xor(sum, 2, 32);
            sum += __shfl_xor(sum, 4, 32);
            sum += __shfl_xor(sum, 8, 32);
            li[r] = li[r] * alpha + sum;
            mi[r] = mnew;
            p0v[r] = e0; p1v[r] = e1;
#pragma unroll
            for (int j = 0; j < 8; j++) o[j][r] *= alpha;
        }

        // D-layout (col-striped) -> A-layout (row-striped) via wave-private LDS
#pragma unroll
        for (int r = 0; r < 8; r++) {
            Pl[wave][r + half8][lanelo]      = (__bf16)p0v[r];
            Pl[wave][r + half8][16 + lanelo] = (__bf16)p1v[r];
        }
        asm volatile("s_wait_dscnt 0" ::: "memory");
        Frag pf;
        pf.u[0] = *(const uint4*)&Pl[wave][lanelo][kb];
        pf.u[1] = *(const uint4*)&Pl[wave][lanelo][kb + 16];

#pragma unroll
        for (int j = 0; j < 8; j++) {
            Frag vf;
            const __bf16* vp = Vb + (size_t)(j * 16 + lanelo) * SS + key0 + kb;
            vf.u[0] = *(const uint4*)vp;
            vf.u[1] = *(const uint4*)(vp + 16);
            o[j] = wmma_bf16(pf, vf, o[j]);
        }
    }

#pragma unroll
    for (int r = 0; r < 8; r++) {
        float inv = 1.0f / li[r];
        int t = q0 + r + half8;
        size_t rowoff = ((size_t)b * SS + t) * DIMSZ + h * HD;
#pragma unroll
        for (int j = 0; j < 8; j++)
            ctx[rowoff + j * 16 + lanelo] = (__bf16)(o[j][r] * inv);
    }
}

// ---------------------------------------------------------------------------
// launch
// ---------------------------------------------------------------------------
extern "C" void kernel_launch(void* const* d_in, const int* in_sizes, int n_in,
                              void* d_out, int out_size, void* d_ws, size_t ws_size,
                              hipStream_t stream) {
    const float* x     = (const float*)d_in[0];
    const float* W_DKV = (const float*)d_in[1];  const float* b_DKV = (const float*)d_in[2];
    const float* W_DQ  = (const float*)d_in[3];  const float* b_DQ  = (const float*)d_in[4];
    const float* W_UK  = (const float*)d_in[5];  const float* b_UK  = (const float*)d_in[6];
    const float* W_UV  = (const float*)d_in[7];  const float* b_UV  = (const float*)d_in[8];
    const float* W_UQ  = (const float*)d_in[9];  const float* b_UQ  = (const float*)d_in[10];
    const float* W_KR  = (const float*)d_in[11]; const float* b_KR  = (const float*)d_in[12];
    const float* W_QR  = (const float*)d_in[13]; const float* b_QR  = (const float*)d_in[14];
    const float* W_O   = (const float*)d_in[15]; const float* b_O   = (const float*)d_in[16];
    float* out = (float*)d_out;

    char* ws = (char*)d_ws;
    size_t off = 0;
    auto alloc = [&](size_t bytes) -> char* {
        char* p = ws + off;
        off += (bytes + 255) & ~(size_t)255;
        return p;
    };
    __bf16* xb     = (__bf16*)alloc((size_t)ROWS * DIMSZ * 2);
    __bf16* WtDKV  = (__bf16*)alloc((size_t)LAT * DIMSZ * 2);
    __bf16* WtDQ   = (__bf16*)alloc((size_t)LAT * DIMSZ * 2);
    __bf16* WtUK   = (__bf16*)alloc((size_t)DIMSZ * LAT * 2);
    __bf16* WtUV   = (__bf16*)alloc((size_t)DIMSZ * LAT * 2);
    __bf16* WtUQ   = (__bf16*)alloc((size_t)DIMSZ * LAT * 2);
    __bf16* WtKR   = (__bf16*)alloc((size_t)HD * DIMSZ * 2);
    __bf16* WtQR   = (__bf16*)alloc((size_t)HD * DIMSZ * 2);
    __bf16* WtO    = (__bf16*)alloc((size_t)DIMSZ * DIMSZ * 2);
    __bf16* ckv    = (__bf16*)alloc((size_t)ROWS * LAT * 2);
    __bf16* cq     = (__bf16*)alloc((size_t)ROWS * LAT * 2);
    __bf16* qcache = (__bf16*)alloc((size_t)BB * NHEAD * SS * 256 * 2);
    __bf16* kcache = (__bf16*)alloc((size_t)BB * NHEAD * SS * 256 * 2);
    __bf16* vT     = (__bf16*)alloc((size_t)BB * NHEAD * HD * SS * 2);
    __bf16* krlin  = (__bf16*)alloc((size_t)ROWS * HD * 2);
    __bf16* qrlin  = (__bf16*)alloc((size_t)ROWS * HD * 2);
    __bf16* ctx    = (__bf16*)alloc((size_t)ROWS * DIMSZ * 2);

    // 1) converts
    {
        int n = ROWS * DIMSZ;
        f32_to_bf16_kernel<<<(n + 255) / 256, 256, 0, stream>>>(x, xb, n);
    }
    auto tr = [&](const float* W, __bf16* Wt, int K, int N) {
        int n = K * N;
        transpose_to_bf16_kernel<<<(n + 255) / 256, 256, 0, stream>>>(W, Wt, K, N);
    };
    tr(W_DKV, WtDKV, DIMSZ, LAT);
    tr(W_DQ,  WtDQ,  DIMSZ, LAT);
    tr(W_UK,  WtUK,  LAT,   DIMSZ);
    tr(W_UV,  WtUV,  LAT,   DIMSZ);
    tr(W_UQ,  WtUQ,  LAT,   DIMSZ);
    tr(W_KR,  WtKR,  DIMSZ, HD);
    tr(W_QR,  WtQR,  DIMSZ, HD);
    tr(W_O,   WtO,   DIMSZ, DIMSZ);

    auto gemm = [&](const __bf16* A, const __bf16* Wt, const float* bias,
                    void* outp, int M, int N, int K, int mode) {
        dim3 grid(N / 128, M / 128);
        gemm_bf16_kernel<<<grid, 256, 0, stream>>>(A, Wt, bias, outp, M, N, K, mode);
    };

    // 2) down projections
    gemm(xb, WtDKV, b_DKV, ckv, ROWS, LAT, DIMSZ, 0);
    gemm(xb, WtDQ,  b_DQ,  cq,  ROWS, LAT, DIMSZ, 0);
    // 3) up projections into head-split caches
    gemm(ckv, WtUK, b_UK, kcache, ROWS, DIMSZ, LAT, 1);
    gemm(ckv, WtUV, b_UV, vT,     ROWS, DIMSZ, LAT, 2);
    gemm(cq,  WtUQ, b_UQ, qcache, ROWS, DIMSZ, LAT, 1);
    // 4) rotary projections + broadcast
    gemm(xb, WtKR, b_KR, krlin, ROWS, HD, DIMSZ, 0);
    gemm(xb, WtQR, b_QR, qrlin, ROWS, HD, DIMSZ, 0);
    {
        int n = ROWS * HD;
        rope_bcast_kernel<<<(n + 255) / 256, 256, 0, stream>>>(krlin, kcache);
        rope_bcast_kernel<<<(n + 255) / 256, 256, 0, stream>>>(qrlin, qcache);
    }
    // 5) attention
    {
        dim3 grid(SS / 128, NHEAD, BB);
        mla_attn_kernel<<<grid, 256, 0, stream>>>(qcache, kcache, vT, ctx);
    }
    // 6) output projection (f32)
    gemm(ctx, WtO, b_O, out, ROWS, DIMSZ, DIMSZ, 3);
}
